// TransformerBlock_56916906606936
// MI455X (gfx1250) — matrix operand order repaired and checked by reading, not compiled, not run on hardware
//
#include <hip/hip_runtime.h>

// =====================================================================
// PointTransformerConv block for MI455X (gfx1250, wave32, WMMA).
//
// Pipeline (all on `stream`):
//   0) zero m/s/acc scratch
//   1) node_qkv: h = LN(relu(x@W_in+b)); q=h@W_dst, k=h@W_src, v=h@W_lin
//      (bf16 WMMA 16x16x32, f32 accumulate; 16-node tile per wave)
//   2) edge pass 1: delta = mlp2(pos[d]-pos[s]); alpha = mlp2(q[d]-k[s]+delta)
//      atomicMax (uint-monotone, alpha>=0) into m[N,64]
//   3) edge pass 2: recompute alpha; e = exp(alpha - m[d]);
//      atomicAdd s += e; atomicAdd acc += e*(v[s]+delta)
//   4) node_out: out = relu((acc/(s+1e-16)) @ W_out + b_out)
//
// Workspace layout (f32): q,k,v,m,s,acc each N*64  (~154 MB total).
// =====================================================================

typedef __attribute__((ext_vector_type(16))) __bf16          v16bf;
typedef __attribute__((ext_vector_type(16))) unsigned short  v16us;
typedef __attribute__((ext_vector_type(8)))  unsigned short  v8us;
typedef __attribute__((ext_vector_type(8)))  float           v8f;

#define TS 72  // padded LDS tile stride (elements); r*TS*elt keeps 16B align

__device__ __forceinline__ unsigned short f2bf(float f) {
  // native fptrunc -> v_cvt_*bf16* (RNE) instead of manual bit twiddling
  return __builtin_bit_cast(unsigned short, (__bf16)f);
}

__device__ __forceinline__ v8f wmma_bf16(v16us a, v16us b, v8f c) {
  return __builtin_amdgcn_wmma_f32_16x16x32_bf16(
      false, __builtin_bit_cast(v16bf, a),
      false, __builtin_bit_cast(v16bf, b),
      (short)0, c, false, false);
}

// B fragment: weights in LDS, bf16, column-major: w[out*64 + k].
// Lane = output column (mod 16); lane half h holds K = {kb*32+h*8 .. +7} in
// elements 0-7 and K = {+16 .. +23} in elements 8-15 (same order as A).
__device__ __forceinline__ v16us bfrag(const unsigned short* w, int nt, int kb, int lane) {
  int col  = nt * 16 + (lane & 15);
  int base = col * 64 + kb * 32 + (lane >> 4) * 8;
  v16us r;
#pragma unroll
  for (int i = 0; i < 8; ++i) { r[i] = w[base + i]; r[i + 8] = w[base + 16 + i]; }
  return r;
}

// A fragment from an f32 row (this lane's row): per ISA layout lane holds
// K = {kb*32+hi*8 .. +7} in VGPRs 0-3 and K = {+16 .. +23} in VGPRs 4-7.
__device__ __forceinline__ v16us afrag_f32(const float* row, int kb, int hi) {
  int c0 = kb * 32 + hi * 8;
  v16us r;
#pragma unroll
  for (int i = 0; i < 8; ++i) {
    r[i]     = f2bf(row[c0 + i]);
    r[i + 8] = f2bf(row[c0 + 16 + i]);
  }
  return r;
}

// A fragment from a bf16 (ushort) row in LDS (already converted).
__device__ __forceinline__ v16us afrag_bf(const unsigned short* row, int kb, int hi) {
  int c0 = kb * 32 + hi * 8;
  v8us u0 = *(const v8us*)(row + c0);
  v8us u1 = *(const v8us*)(row + c0 + 16);
  v16us r;
#pragma unroll
  for (int i = 0; i < 8; ++i) { r[i] = u0[i]; r[i + 8] = u1[i]; }
  return r;
}

// ---------------------------------------------------------------------
__global__ void zero_kernel(float* __restrict__ p, long n) {
  long i = (long)blockIdx.x * blockDim.x + threadIdx.x;
  long stride = (long)gridDim.x * blockDim.x;
  for (; i < n; i += stride) p[i] = 0.f;
}

// ---------------------------------------------------------------------
// Node kernel: h = LN(relu(x@W_in+b_in)); q = h@W_dst; k = h@W_src; v = h@W_lin
__global__ __launch_bounds__(128) void node_qkv_kernel(
    const float* __restrict__ x, const float* __restrict__ Win, const float* __restrict__ bin,
    const float* __restrict__ lng, const float* __restrict__ lnb,
    const float* __restrict__ Wlin, const float* __restrict__ Wsrc, const float* __restrict__ Wdst,
    float* __restrict__ q, float* __restrict__ k, float* __restrict__ v, int N) {
  __shared__ __align__(16) unsigned short s_Win[4096], s_Wlin[4096], s_Wsrc[4096], s_Wdst[4096];
  __shared__ float s_bin[64], s_g[64], s_bb[64];
  __shared__ __align__(16) float          s_h [4][16 * TS];
  __shared__ __align__(16) unsigned short s_hn[4][16 * TS];

  const int tid = threadIdx.x;
  for (int t = tid; t < 4096; t += 128) {
    int o = t >> 6, i = t & 63;                 // col-major bf16: [o*64 + i]
    s_Win [t] = f2bf(Win [i * 64 + o]);
    s_Wlin[t] = f2bf(Wlin[i * 64 + o]);
    s_Wsrc[t] = f2bf(Wsrc[i * 64 + o]);
    s_Wdst[t] = f2bf(Wdst[i * 64 + o]);
  }
  if (tid < 64) { s_bin[tid] = bin[tid]; s_g[tid] = lng[tid]; s_bb[tid] = lnb[tid]; }
  __syncthreads();

  const int wave = tid >> 5, lane = tid & 31, hi = lane >> 4, r = lane & 15;
  const long base = ((long)blockIdx.x * 4 + wave) * 16;
  long nid = base + r; if (nid >= N) nid = N - 1;   // clamp; stores are masked
  const float* xrow = x + nid * 64;
  __builtin_prefetch(xrow, 0, 1);

  v16us xA[2];
  xA[0] = afrag_f32(xrow, 0, hi);
  xA[1] = afrag_f32(xrow, 1, hi);

  float* th = s_h[wave];
#pragma unroll
  for (int nt = 0; nt < 4; ++nt) {
    v8f acc = {};
    acc = wmma_bf16(xA[0], bfrag(s_Win, nt, 0, lane), acc);
    acc = wmma_bf16(xA[1], bfrag(s_Win, nt, 1, lane), acc);
    int col = nt * 16 + r;
    float b = s_bin[col];
#pragma unroll
    for (int vv = 0; vv < 8; ++vv)
      th[(vv + hi * 8) * TS + col] = fmaxf(acc[vv] + b, 0.f);  // relu
  }
  __syncthreads();

  // LayerNorm: lane owns row r, channels [hi*32, hi*32+32); halves combine via shfl_xor(16)
  {
    const int cb = hi * 32;
    float sm = 0.f, sq = 0.f;
#pragma unroll
    for (int c = 0; c < 32; ++c) { float t = th[r * TS + cb + c]; sm += t; sq += t * t; }
    sm += __shfl_xor(sm, 16, 32);
    sq += __shfl_xor(sq, 16, 32);
    float mu  = sm * (1.f / 64.f);
    float var = sq * (1.f / 64.f) - mu * mu;
    float inv = rsqrtf(var + 1e-5f);
    unsigned short* thn = s_hn[wave];
#pragma unroll
    for (int c = 0; c < 32; ++c) {
      int cc = cb + c;
      float t = th[r * TS + cc];
      thn[r * TS + cc] = f2bf((t - mu) * inv * s_g[cc] + s_bb[cc]);
    }
  }
  __syncthreads();

  v16us hA[2];
  hA[0] = afrag_bf(s_hn[wave] + r * TS, 0, hi);
  hA[1] = afrag_bf(s_hn[wave] + r * TS, 1, hi);

  // q = h@W_dst ; k = h@W_src ; v = h@W_lin
#pragma unroll
  for (int m_ = 0; m_ < 3; ++m_) {
    const unsigned short* W = (m_ == 0) ? s_Wdst : (m_ == 1) ? s_Wsrc : s_Wlin;
    float* outp             = (m_ == 0) ? q      : (m_ == 1) ? k      : v;
#pragma unroll
    for (int nt = 0; nt < 4; ++nt) {
      v8f acc = {};
      acc = wmma_bf16(hA[0], bfrag(W, nt, 0, lane), acc);
      acc = wmma_bf16(hA[1], bfrag(W, nt, 1, lane), acc);
      int col = nt * 16 + r;
#pragma unroll
      for (int vv = 0; vv < 8; ++vv) {
        long row = base + vv + hi * 8;
        if (row < N) outp[row * 64 + col] = acc[vv];
      }
    }
  }
}

// ---------------------------------------------------------------------
// Edge kernels. PASS=1: atomicMax alpha into m. PASS=2: accumulate s/acc.
template <int PASS>
__global__ __launch_bounds__(128) void edge_pass_kernel(
    const float* __restrict__ pos, const int* __restrict__ ei,
    const float* __restrict__ q, const float* __restrict__ k, const float* __restrict__ v,
    const float* __restrict__ pnW1, const float* __restrict__ pnb1,
    const float* __restrict__ pnW2, const float* __restrict__ pnb2,
    const float* __restrict__ anW1, const float* __restrict__ anb1,
    const float* __restrict__ anW2, const float* __restrict__ anb2,
    float* __restrict__ mbuf, float* __restrict__ sbuf, float* __restrict__ abuf,
    int E) {
  __shared__ __align__(16) unsigned short s_pnW2[4096], s_anW1[4096], s_anW2[4096];
  __shared__ float s_pW1[3 * 64];
  __shared__ float s_b[4][64];                 // pnb1, pnb2, anb1, anb2
  __shared__ __align__(16) float s_tA[4][16 * TS];   // delta (kept for pass 2 output)
  __shared__ __align__(16) float s_tB[4][16 * TS];   // hidden / alpha scratch

  const int tid = threadIdx.x;
  for (int t = tid; t < 4096; t += 128) {
    int o = t >> 6, i = t & 63;
    s_pnW2[t] = f2bf(pnW2[i * 64 + o]);
    s_anW1[t] = f2bf(anW1[i * 64 + o]);
    s_anW2[t] = f2bf(anW2[i * 64 + o]);
  }
  for (int t = tid; t < 192; t += 128) s_pW1[t] = pnW1[t];
  if (tid < 64) {
    s_b[0][tid] = pnb1[tid]; s_b[1][tid] = pnb2[tid];
    s_b[2][tid] = anb1[tid]; s_b[3][tid] = anb2[tid];
  }
  __syncthreads();

  const int wave = tid >> 5, lane = tid & 31, hi = lane >> 4, r = lane & 15;
  const long ebase = ((long)blockIdx.x * 4 + wave) * 16;
  long eidx = ebase + r; if (eidx >= E) eidx = E - 1;     // clamp; scatter masked
  const int si = ei[eidx];                 // source j
  const int di = ei[(long)E + eidx];       // target i

  const float* qrow = q + (long)di * 64;
  const float* krow = k + (long)si * 64;
  __builtin_prefetch(qrow, 0, 1);
  __builtin_prefetch(krow, 0, 1);

  // pos_nn layer 1 (3 -> 64) + relu, built directly in WMMA-A layout
  const float dx = pos[(long)di * 3 + 0] - pos[(long)si * 3 + 0];
  const float dy = pos[(long)di * 3 + 1] - pos[(long)si * 3 + 1];
  const float dz = pos[(long)di * 3 + 2] - pos[(long)si * 3 + 2];
  v16us hA[2];
#pragma unroll
  for (int kb = 0; kb < 2; ++kb)
#pragma unroll
    for (int half = 0; half < 2; ++half)
#pragma unroll
      for (int j = 0; j < 8; ++j) {
        int c = kb * 32 + half * 16 + hi * 8 + j;
        float hv = s_b[0][c] + dx * s_pW1[c] + dy * s_pW1[64 + c] + dz * s_pW1[128 + c];
        hA[kb][half * 8 + j] = f2bf(fmaxf(hv, 0.f));
      }

  float* tA = s_tA[wave];
  float* tB = s_tB[wave];

  // pos_nn layer 2 (64 -> 64) + relu -> delta, store f32 tile
#pragma unroll
  for (int nt = 0; nt < 4; ++nt) {
    v8f acc = {};
    acc = wmma_bf16(hA[0], bfrag(s_pnW2, nt, 0, lane), acc);
    acc = wmma_bf16(hA[1], bfrag(s_pnW2, nt, 1, lane), acc);
    int col = nt * 16 + r;
    float b2 = s_b[1][col];
#pragma unroll
    for (int vv = 0; vv < 8; ++vv)
      tA[(vv + hi * 8) * TS + col] = fmaxf(acc[vv] + b2, 0.f);
  }
  __syncthreads();

  // attn input: q[dst] - k[src] + delta -> A frags
  v16us aA[2];
#pragma unroll
  for (int kb = 0; kb < 2; ++kb) {
    int c0 = kb * 32 + hi * 8;
#pragma unroll
    for (int j = 0; j < 8; ++j) {
      float x0 = qrow[c0 + j]      - krow[c0 + j]      + tA[r * TS + c0 + j];
      float x1 = qrow[c0 + 16 + j] - krow[c0 + 16 + j] + tA[r * TS + c0 + 16 + j];
      aA[kb][j]     = f2bf(x0);
      aA[kb][j + 8] = f2bf(x1);
    }
  }

  // attn_nn layer 1 + relu -> tile B
#pragma unroll
  for (int nt = 0; nt < 4; ++nt) {
    v8f acc = {};
    acc = wmma_bf16(aA[0], bfrag(s_anW1, nt, 0, lane), acc);
    acc = wmma_bf16(aA[1], bfrag(s_anW1, nt, 1, lane), acc);
    int col = nt * 16 + r;
    float b1 = s_b[2][col];
#pragma unroll
    for (int vv = 0; vv < 8; ++vv)
      tB[(vv + hi * 8) * TS + col] = fmaxf(acc[vv] + b1, 0.f);
  }
  __syncthreads();

  v16us gA[2];
  gA[0] = afrag_f32(tB + r * TS, 0, hi);
  gA[1] = afrag_f32(tB + r * TS, 1, hi);
  __syncthreads();  // everyone done reading tB before it is overwritten

  // attn_nn layer 2 + relu -> alpha, into tile B
#pragma unroll
  for (int nt = 0; nt < 4; ++nt) {
    v8f acc = {};
    acc = wmma_bf16(gA[0], bfrag(s_anW2, nt, 0, lane), acc);
    acc = wmma_bf16(gA[1], bfrag(s_anW2, nt, 1, lane), acc);
    int col = nt * 16 + r;
    float b2 = s_b[3][col];
#pragma unroll
    for (int vv = 0; vv < 8; ++vv)
      tB[(vv + hi * 8) * TS + col] = fmaxf(acc[vv] + b2, 0.f);
  }
  __syncthreads();

  // Scatter: lane owns edge row r, channels [hi*32, +32). alpha >= 0, so
  // float atomic max == uint atomic max on the bit pattern (m initialized 0).
  const bool vrow = (ebase + r) < (long)E;
  const int cb = hi * 32;
  if (PASS == 1) {
    if (vrow) {
      unsigned* mrow = (unsigned*)(mbuf + (long)di * 64 + cb);
#pragma unroll
      for (int c = 0; c < 32; ++c)
        atomicMax(mrow + c, __float_as_uint(tB[r * TS + cb + c]));
    }
  } else {
    if (vrow) {
      const float* vrowp = v + (long)si * 64 + cb;
      const float* mrow  = mbuf + (long)di * 64 + cb;
      float* srowp = sbuf + (long)di * 64 + cb;
      float* arowp = abuf + (long)di * 64 + cb;
      __builtin_prefetch(vrowp, 0, 1);
#pragma unroll
      for (int c = 0; c < 32; ++c) {
        float al = tB[r * TS + cb + c];
        float ee = __expf(al - mrow[c]);
        atomicAdd(srowp + c, ee);
        atomicAdd(arowp + c, ee * (vrowp[c] + tA[r * TS + cb + c]));
      }
    }
  }
}

// ---------------------------------------------------------------------
// Final: out = relu((acc / (s + 1e-16)) @ W_out + b_out)
__global__ __launch_bounds__(128) void node_out_kernel(
    const float* __restrict__ abuf, const float* __restrict__ sbuf,
    const float* __restrict__ Wout, const float* __restrict__ bout,
    float* __restrict__ out, int N) {
  __shared__ __align__(16) unsigned short s_W[4096];
  __shared__ float s_b[64];
  __shared__ __align__(16) unsigned short s_y[4][16 * TS];

  const int tid = threadIdx.x;
  for (int t = tid; t < 4096; t += 128) {
    int o = t >> 6, i = t & 63;
    s_W[t] = f2bf(Wout[i * 64 + o]);
  }
  if (tid < 64) s_b[tid] = bout[tid];
  __syncthreads();

  const int wave = tid >> 5, lane = tid & 31, hi = lane >> 4, r = lane & 15;
  const long base = ((long)blockIdx.x * 4 + wave) * 16;
  long nid = base + r; if (nid >= N) nid = N - 1;
  const int cb = hi * 32;
  {
    const float* ar = abuf + nid * 64 + cb;
    const float* sr = sbuf + nid * 64 + cb;
    unsigned short* ty = s_y[wave];
#pragma unroll
    for (int c = 0; c < 32; ++c)
      ty[r * TS + cb + c] = f2bf(ar[c] / (sr[c] + 1e-16f));
  }
  __syncthreads();

  v16us yA[2];
  yA[0] = afrag_bf(s_y[wave] + r * TS, 0, hi);
  yA[1] = afrag_bf(s_y[wave] + r * TS, 1, hi);

#pragma unroll
  for (int nt = 0; nt < 4; ++nt) {
    v8f acc = {};
    acc = wmma_bf16(yA[0], bfrag(s_W, nt, 0, lane), acc);
    acc = wmma_bf16(yA[1], bfrag(s_W, nt, 1, lane), acc);
    int col = nt * 16 + r;
    float b = s_b[col];
#pragma unroll
    for (int vv = 0; vv < 8; ++vv) {
      long row = base + vv + hi * 8;
      if (row < N) out[row * 64 + col] = fmaxf(acc[vv] + b, 0.f);
    }
  }
}

// ---------------------------------------------------------------------
extern "C" void kernel_launch(void* const* d_in, const int* in_sizes, int n_in,
                              void* d_out, int out_size, void* d_ws, size_t ws_size,
                              hipStream_t stream) {
  const float* x    = (const float*)d_in[0];
  const float* pos  = (const float*)d_in[1];
  const int*   ei   = (const int*)  d_in[2];
  const float* Win  = (const float*)d_in[3];
  const float* bin  = (const float*)d_in[4];
  const float* lng  = (const float*)d_in[5];
  const float* lnb  = (const float*)d_in[6];
  const float* Wlin = (const float*)d_in[7];
  const float* Wsrc = (const float*)d_in[8];
  const float* Wdst = (const float*)d_in[9];
  const float* pnW1 = (const float*)d_in[10];
  const float* pnb1 = (const float*)d_in[11];
  const float* pnW2 = (const float*)d_in[12];
  const float* pnb2 = (const float*)d_in[13];
  const float* anW1 = (const float*)d_in[14];
  const float* anb1 = (const float*)d_in[15];
  const float* anW2 = (const float*)d_in[16];
  const float* anb2 = (const float*)d_in[17];
  const float* Wout = (const float*)d_in[18];
  const float* bout = (const float*)d_in[19];

  const int N = in_sizes[0] / 64;
  const int E = in_sizes[2] / 2;

  float* ws = (float*)d_ws;
  const size_t st = (size_t)N * 64;
  float* qb = ws;            // q
  float* kb = ws + st;       // k
  float* vb = ws + 2 * st;   // v
  float* mb = ws + 3 * st;   // per (dst,chan) max
  float* sb = ws + 4 * st;   // per (dst,chan) sum of exp
  float* ab = ws + 5 * st;   // per (dst,chan) weighted accumulation

  // 0) zero m, s, acc (contiguous 3*st floats starting at mb)
  zero_kernel<<<dim3(2048), dim3(256), 0, stream>>>(mb, (long)(3 * st));

  const int ntiles  = (N + 15) / 16;
  const int nblocks = (ntiles + 3) / 4;
  node_qkv_kernel<<<dim3(nblocks), dim3(128), 0, stream>>>(
      x, Win, bin, lng, lnb, Wlin, Wsrc, Wdst, qb, kb, vb, N);

  const int etiles  = (E + 15) / 16;
  const int eblocks = (etiles + 3) / 4;
  edge_pass_kernel<1><<<dim3(eblocks), dim3(128), 0, stream>>>(
      pos, ei, qb, kb, vb, pnW1, pnb1, pnW2, pnb2, anW1, anb1, anW2, anb2,
      mb, sb, ab, E);
  edge_pass_kernel<2><<<dim3(eblocks), dim3(128), 0, stream>>>(
      pos, ei, qb, kb, vb, pnW1, pnb1, pnW2, pnb2, anW1, anb1, anW2, anb2,
      mb, sb, ab, E);

  node_out_kernel<<<dim3(nblocks), dim3(128), 0, stream>>>(
      ab, sb, Wout, bout, (float*)d_out, N);
}
